// Generator_1434519077124
// MI455X (gfx1250) — matrix operand order repaired and checked by reading, not compiled, not run on hardware
//
#include <hip/hip_runtime.h>
#include <math.h>

typedef __attribute__((ext_vector_type(16))) _Float16 v16h;
typedef __attribute__((ext_vector_type(8)))  _Float16 v8h;
typedef __attribute__((ext_vector_type(8)))  float    v8f;

#define SQRT2F 1.41421356237309515f

__device__ __forceinline__ float flrelu1(float v) {
    return (v > 0.f ? v : 0.2f * v) * SQRT2F;
}
__device__ __forceinline__ float blurw(int a) {  // [1,3,3,1]
    return (a == 0 || a == 3) ? 1.f : 3.f;
}
__device__ __forceinline__ v16h cat8(v8h lo, v8h hi) {
    return __builtin_shufflevector(lo, hi, 0, 1, 2, 3, 4, 5, 6, 7,
                                   8, 9, 10, 11, 12, 13, 14, 15);
}

// ---------------------------------------------------------------- mapping ---
__global__ void pixelnorm_kernel(const float* __restrict__ z, float* __restrict__ out) {
    __shared__ float red[256];
    int r = blockIdx.x, t = threadIdx.x;
    float a = z[r * 512 + t];
    float b = z[r * 512 + 256 + t];
    red[t] = a * a + b * b;
    __syncthreads();
    for (int s = 128; s > 0; s >>= 1) {
        if (t < s) red[t] += red[t + s];
        __syncthreads();
    }
    float inv = rsqrtf(red[0] * (1.f / 512.f) + 1e-8f);
    out[r * 512 + t] = a * inv;
    out[r * 512 + 256 + t] = b * inv;
}

__global__ void eqlin_lrelu_kernel(const float* __restrict__ x, const float* __restrict__ w,
                                   const float* __restrict__ bias, float* __restrict__ out,
                                   float wscale, float bmul) {
    int idx = blockIdx.x * blockDim.x + threadIdx.x;  // 4*512
    int r = idx >> 9, o = idx & 511;
    const float* xr = x + r * 512;
    const float* wr = w + o * 512;
    float acc = 0.f;
    for (int c = 0; c < 512; ++c) acc += xr[c] * wr[c];
    out[idx] = flrelu1(acc * wscale + bias[o] * bmul);
}

// ------------------------------------------------------- style / demod ------
__global__ void style_kernel(const float* __restrict__ wlat, const float* __restrict__ mod_w,
                             const float* __restrict__ mod_b, float* __restrict__ s, int Cin) {
    int idx = blockIdx.x * blockDim.x + threadIdx.x;
    if (idx >= 4 * Cin) return;
    int b = idx / Cin, c = idx - b * Cin;
    const float* wr = wlat + b * 512;
    const float* mw = mod_w + (size_t)c * 512;
    float acc = 0.f;
    for (int k = 0; k < 512; ++k) acc += wr[k] * mw[k];
    s[idx] = acc * 0.04419417382415922f /*1/sqrt(512)*/ + mod_b[c];
}

__global__ void wsq_kernel(const float* __restrict__ W, float* __restrict__ wsq, int total) {
    int idx = blockIdx.x * blockDim.x + threadIdx.x;
    if (idx >= total) return;
    const float* p = W + (size_t)idx * 9;
    float a = 0.f;
    for (int i = 0; i < 9; ++i) a += p[i] * p[i];
    wsq[idx] = a;
}

__global__ void demod_kernel(const float* __restrict__ s, const float* __restrict__ wsq,
                             float* __restrict__ d, int Cin, int Cout, float scale2) {
    int idx = blockIdx.x * blockDim.x + threadIdx.x;
    if (idx >= 4 * Cout) return;
    int b = idx / Cout, co = idx - b * Cout;
    const float* sb = s + b * Cin;
    const float* wq = wsq + (size_t)co * Cin;
    float acc = 0.f;
    for (int c = 0; c < Cin; ++c) {
        float sv = sb[c];
        acc += sv * sv * wq[c];
    }
    d[idx] = rsqrtf(acc * scale2 + 1e-8f);
}

// dense weight image for mode-0 convs: a16[cout][cin*9 + 3*dy+dx]
__global__ void wcvt_kernel(const float* __restrict__ W, _Float16* __restrict__ a16,
                            int total, float scale) {
    int idx = blockIdx.x * blockDim.x + threadIdx.x;
    if (idx >= total) return;
    a16[idx] = (_Float16)(W[idx] * scale);
}

// phase-packed flipped weights for transposed conv.
// phase (py,px): p-taps = py?{1}:{0,2}, q-taps = px?{1}:{0,2};
// A_ph[cout][cin*ntap + t] = scale * W[cout][cin][2-p][2-q]
__global__ void wcvt_phase_kernel(const float* __restrict__ W, _Float16* __restrict__ a16,
                                  int Cin, int total, int py, int px, float scale) {
    int idx = blockIdx.x * blockDim.x + threadIdx.x;
    if (idx >= total) return;  // total = Cout*Cin*ntap
    int nq = px ? 1 : 2;
    int np = py ? 1 : 2;
    int ntap = np * nq;
    int t = idx % ntap;
    int c1 = idx / ntap;  // cout*Cin + cin
    int pp = py ? 1 : 2 * (t / nq);
    int qq = px ? 1 : 2 * (t % nq);
    int r = (2 - pp) * 3 + (2 - qq);
    a16[idx] = (_Float16)(W[(size_t)c1 * 9 + r] * scale);
}

__global__ void modulate_kernel(const float* __restrict__ act, const float* __restrict__ s,
                                _Float16* __restrict__ xm, int Cin, int HW, int bstride,
                                int total) {
    int idx = blockIdx.x * blockDim.x + threadIdx.x;
    if (idx >= total) return;
    int hw = idx % HW;
    int t1 = idx / HW;
    int c = t1 % Cin, b = t1 / Cin;
    float v = act[(size_t)b * bstride + (size_t)c * HW + hw] * s[b * Cin + c];
    xm[idx] = (_Float16)v;
}

// ------------------------------------------------ WMMA implicit-GEMM conv ---
// Block tile 128(Cout) x 64(pixels); 8 waves as 4(M) x 2(N); each wave owns a
// 32x32 output slab -> 4 WMMAs per K-step reusing 2 A and 2 B fragments.
// mode 0: normal 3x3 conv, pad 1, K = 9*Cin; epilogue demod+bias+fLReLU into
//         channels [0,Cout) of [B,2*Cout,H,H].
// mode 1: one parity phase (py,px) of the transposed conv; K = ntap*Cin with
//         ntap in {1,2,4} (phase-packed flipped weights); raw result is
//         scattered into the (2H+1)^2 convT grid.
__global__ void __launch_bounds__(256)
conv_wmma_kernel(const _Float16* __restrict__ a16, const _Float16* __restrict__ xm,
                 const float* __restrict__ d, const float* __restrict__ act_b,
                 float* __restrict__ out, int Cin, int Cout, int H, int mode,
                 int py, int px) {
    const int b = blockIdx.z;
    const int m0 = blockIdx.y * 128;
    const int n0 = blockIdx.x * 64;

    int Ktot, Npix, Wo = 1, Wp = 1, ntap = 1, nq = 1;
    if (mode == 0) {
        Wo = H; Npix = H * H; Ktot = Cin * 9;
    } else {
        int np = py ? 1 : 2;
        nq = px ? 1 : 2;
        ntap = np * nq;
        Ktot = Cin * ntap;
        int Hp = py ? H : H + 1;
        Wp = px ? H : H + 1;
        Npix = Hp * Wp;
    }

    __shared__ _Float16 sA[128][32];  // [m][k]  rows 64B
    __shared__ _Float16 sBt[64][40];  // [n][k]  rows 80B (16B-aligned chunks)

    const int t = threadIdx.x;
    const int lane = t & 31, wave = t >> 5;
    const int wm = wave >> 1, wn = wave & 1;
    const int hi = lane >> 4;

    v8f acc00 = {}, acc01 = {}, acc10 = {}, acc11 = {};
    const _Float16* xb = xm + (size_t)b * Cin * H * H;

    const int am = t >> 1, ak = (t & 1) * 16;  // A staging: 2x v8h per thread
    const int bk = t >> 3, bn0 = (t & 7) * 8;  // B staging: 8 pixels at fixed k

    const int mr0 = wm * 32 + (lane & 15);
    const int mr1 = mr0 + 16;
    const int nc0 = wn * 32 + (lane & 15);
    const int nc1 = nc0 + 16;

    for (int k0 = 0; k0 < Ktot; k0 += 32) {
        {  // stage A: global_load_b128 x2 + ds_store_b128 x2
            const _Float16* src = a16 + (size_t)(m0 + am) * Ktot + k0 + ak;
            *(v8h*)&sA[am][ak] = *(const v8h*)src;
            *(v8h*)&sA[am][ak + 8] = *(const v8h*)(src + 8);
            if (k0 + 32 < Ktot)
                __builtin_prefetch(src + 32, 0, 1);  // global_prefetch next K chunk
        }
        {  // stage B: im2col gather, transposed into sBt[n][k]
            int kg = k0 + bk;
            int cin, dy = 0, dx = 0, pp = 0, qq = 0;
            if (mode == 0) {
                cin = kg / 9;
                int r = kg - cin * 9;
                dy = r / 3;
                dx = r - dy * 3;
            } else {
                cin = kg / ntap;
                int tt = kg - cin * ntap;
                pp = py ? 1 : 2 * (tt / nq);
                qq = px ? 1 : 2 * (tt % nq);
            }
            const _Float16* plane = xb + (size_t)cin * H * H;
#pragma unroll
            for (int i = 0; i < 8; ++i) {
                int n = bn0 + i;
                int p = n0 + n;
                _Float16 v = (_Float16)0.f;
                if (p < Npix) {
                    if (mode == 0) {
                        int y = p / Wo, x = p - y * Wo;
                        int sy = y + dy - 1, sx = x + dx - 1;
                        if ((unsigned)sy < (unsigned)H && (unsigned)sx < (unsigned)H)
                            v = plane[sy * H + sx];
                    } else {
                        int yy = p / Wp, xx = p - yy * Wp;
                        int u = 2 * yy + py + pp, w = 2 * xx + px + qq;
                        if (u >= 2 && u <= 2 * H && w >= 2 && w <= 2 * H)
                            v = plane[((u >> 1) - 1) * H + ((w >> 1) - 1)];
                    }
                }
                sBt[n][bk] = v;
            }
        }
        __syncthreads();
        // Fragments per ISA 7.12.2: per lane, A = two contiguous 8-half runs
        // (k = hi*8.., 16+hi*8..); B (transposed in LDS) = same runs (k = hi*8.., 16+hi*8..)
        const v8h* ar0 = (const v8h*)&sA[mr0][0];
        v16h af0 = cat8(ar0[hi], ar0[2 + hi]);
        const v8h* ar1 = (const v8h*)&sA[mr1][0];
        v16h af1 = cat8(ar1[hi], ar1[2 + hi]);
        const _Float16* br0 = &sBt[nc0][0];
        v16h bf0 = cat8(*(const v8h*)(br0 + hi * 8), *(const v8h*)(br0 + 16 + hi * 8));
        const _Float16* br1 = &sBt[nc1][0];
        v16h bf1 = cat8(*(const v8h*)(br1 + hi * 8), *(const v8h*)(br1 + 16 + hi * 8));

        acc00 = __builtin_amdgcn_wmma_f32_16x16x32_f16(false, af0, false, bf0,
                                                       (short)0, acc00, false, false);
        acc01 = __builtin_amdgcn_wmma_f32_16x16x32_f16(false, af0, false, bf1,
                                                       (short)0, acc01, false, false);
        acc10 = __builtin_amdgcn_wmma_f32_16x16x32_f16(false, af1, false, bf0,
                                                       (short)0, acc10, false, false);
        acc11 = __builtin_amdgcn_wmma_f32_16x16x32_f16(false, af1, false, bf1,
                                                       (short)0, acc11, false, false);
        __syncthreads();
    }

    // epilogue; C/D layout: VGPR r -> M = r + 8*hi, N = lane&15
    const int Ho = 2 * H + 1;
#pragma unroll
    for (int mi = 0; mi < 2; ++mi) {
#pragma unroll
        for (int r = 0; r < 8; ++r) {
            int cout = m0 + wm * 32 + mi * 16 + hi * 8 + r;
            float dm = 0.f, ab = 0.f;
            if (mode == 0) { dm = d[b * Cout + cout]; ab = act_b[cout]; }
#pragma unroll
            for (int h = 0; h < 2; ++h) {
                int col = n0 + (h ? nc1 : nc0);
                if (col >= Npix) continue;
                float v = mi ? (h ? acc11[r] : acc10[r]) : (h ? acc01[r] : acc00[r]);
                if (mode == 0) {
                    out[((size_t)b * 2 * Cout + cout) * Npix + col] = flrelu1(v * dm + ab);
                } else {
                    int yy = col / Wp, xx = col - yy * Wp;
                    int yo = 2 * yy + py, xo = 2 * xx + px;
                    out[((size_t)b * Cout + cout) * Ho * Ho + yo * Ho + xo] = v;
                }
            }
        }
    }
}

// --------------------------------------- blur (4x4 FIR) after up-conv ------
__global__ void blur_act_kernel(const float* __restrict__ convT, const float* __restrict__ d,
                                const float* __restrict__ act_b, float* __restrict__ out,
                                int Cout, int H, int total) {
    int idx = blockIdx.x * blockDim.x + threadIdx.x;
    if (idx >= total) return;  // total = 4*Cout*H*H
    int x = idx % H;
    int t1 = idx / H;
    int y = t1 % H;
    int t2 = t1 / H;
    int c = t2 % Cout, b = t2 / Cout;
    int Ho = H + 1;
    const float* src = convT + ((size_t)b * Cout + c) * Ho * Ho;
    float acc = 0.f;
    for (int a = 0; a < 4; ++a) {
        int u = y - 1 + a;
        if ((unsigned)u >= (unsigned)Ho) continue;
        float ka = blurw(a);
        for (int bb = 0; bb < 4; ++bb) {
            int v = x - 1 + bb;
            if ((unsigned)v >= (unsigned)Ho) continue;
            acc += ka * blurw(bb) * src[u * Ho + v];
        }
    }
    acc *= (1.f / 16.f);  // BLUR_K * 4  ==  outer([1,3,3,1]) / 16
    acc = acc * d[b * Cout + c] + act_b[c];
    out[((size_t)b * 2 * Cout + c) * H * H + y * H + x] = flrelu1(acc);
}

// ----------------------------------------------- noise concat half ---------
__global__ void noise_fill_kernel(const float* __restrict__ noise, const float* __restrict__ nw,
                                  const float* __restrict__ act_b, float* __restrict__ out,
                                  int Cout, int HW, int total) {
    int idx = blockIdx.x * blockDim.x + threadIdx.x;
    if (idx >= total) return;  // total = 4*Cout*HW, noise is [B,Cout,HW]
    int i = idx % HW;
    int t1 = idx / HW;
    int c = t1 % Cout, b = t1 / Cout;
    float v = (*nw) * noise[idx] + act_b[Cout + c];
    out[((size_t)b * 2 * Cout + Cout + c) * HW + i] = flrelu1(v);
}

// --------------------------------------------------------- toRGB -----------
__global__ void wrgb_kernel(const float* __restrict__ W, const float* __restrict__ s,
                            float* __restrict__ wrgb, int Cin, float scale, int total) {
    int idx = blockIdx.x * blockDim.x + threadIdx.x;
    if (idx >= total) return;  // total = 4*3*Cin
    int cin = idx % Cin;
    int t1 = idx / Cin;
    int co = t1 % 3, b = t1 / 3;
    wrgb[idx] = W[co * Cin + cin] * s[b * Cin + cin] * scale;
}

__global__ void rgb_kernel(const float* __restrict__ act, const float* __restrict__ wrgb,
                           const float* __restrict__ bias, float* __restrict__ out,
                           int Cin, int HW, int total) {
    int idx = blockIdx.x * blockDim.x + threadIdx.x;
    if (idx >= total) return;  // total = 4*HW
    int p = idx % HW;
    int b = idx / HW;
    const float* a = act + (size_t)b * Cin * HW + p;
    const float* w0 = wrgb + (size_t)b * 3 * Cin;
    float r = 0.f, g = 0.f, bl = 0.f;
    for (int c = 0; c < Cin; ++c) {
        float v = a[(size_t)c * HW];
        r += w0[c] * v;
        g += w0[Cin + c] * v;
        bl += w0[2 * Cin + c] * v;
    }
    out[((size_t)b * 3 + 0) * HW + p] = r + bias[0];
    out[((size_t)b * 3 + 1) * HW + p] = g + bias[1];
    out[((size_t)b * 3 + 2) * HW + p] = bl + bias[2];
}

// skip upsample (upfirdn up=2, pad=(2,1), kernel BLUR_K*4), accumulate into out
__global__ void skip_up_add_kernel(const float* __restrict__ skip, float* __restrict__ out,
                                   int Hin, int total) {
    int idx = blockIdx.x * blockDim.x + threadIdx.x;
    if (idx >= total) return;  // total = 4*3*(2*Hin)^2
    int Hout = 2 * Hin;
    int x = idx % Hout;
    int t1 = idx / Hout;
    int y = t1 % Hout;
    int bc = t1 / Hout;  // [0,12)
    const float* src = skip + (size_t)bc * Hin * Hin;
    float acc = 0.f;
    for (int a = 0; a < 4; ++a) {
        int u = y + a - 2;
        if (u < 0 || (u & 1)) continue;
        u >>= 1;
        if (u >= Hin) continue;
        float ka = blurw(a);
        for (int bb = 0; bb < 4; ++bb) {
            int v = x + bb - 2;
            if (v < 0 || (v & 1)) continue;
            v >>= 1;
            if (v >= Hin) continue;
            acc += ka * blurw(bb) * src[u * Hin + v];
        }
    }
    out[idx] += acc * (1.f / 16.f);
}

// ===========================================================================
struct CCfg { int cin, cout, Hin, up; };
static const CCfg g_conv[11] = {
    {512, 512, 4, 0},
    {1024, 512, 4, 1},  {1024, 512, 8, 0},
    {1024, 512, 8, 1},  {1024, 512, 16, 0},
    {1024, 512, 16, 1}, {1024, 512, 32, 0},
    {1024, 512, 32, 1}, {1024, 512, 64, 0},
    {1024, 256, 64, 1}, {512, 256, 128, 0},
};
static const int g_rgb_cin[6] = {1024, 1024, 1024, 1024, 1024, 512};

static inline int cdiv(int a, int b) { return (a + b - 1) / b; }

extern "C" void kernel_launch(void* const* d_in, const int* in_sizes, int n_in,
                              void* d_out, int out_size, void* d_ws, size_t ws_size,
                              hipStream_t stream) {
    (void)in_sizes; (void)n_in; (void)out_size; (void)ws_size;
    const float* z = (const float*)d_in[0];
    const float* cst = (const float*)d_in[1];

    // ---- workspace carve-out (~620 MB) ----
    char* ws = (char*)d_ws;
    size_t off = 0;
    auto carve = [&](size_t bytes) -> void* {
        off = (off + 255) & ~(size_t)255;
        void* p = ws + off;
        off += bytes;
        return p;
    };
    float*     wlat  = (float*)carve(4 * 512 * 4);
    float*     wtmp  = (float*)carve(4 * 512 * 4);
    float*     sbuf  = (float*)carve(4 * 1024 * 4);
    float*     dbuf  = (float*)carve(4 * 512 * 4);
    float*     wsqb  = (float*)carve((size_t)512 * 1024 * 4);
    float*     wrgbb = (float*)carve(4 * 3 * 1024 * 4);
    _Float16*  a16   = (_Float16*)carve((size_t)512 * 1024 * 9 * 2);
    _Float16*  xm16  = (_Float16*)carve((size_t)4 * 512 * 128 * 128 * 2);
    float*     actA  = (float*)carve((size_t)4 * 512 * 128 * 128 * 4);
    float*     actB  = (float*)carve((size_t)4 * 512 * 128 * 128 * 4);
    float*     convT = (float*)carve((size_t)4 * 256 * 257 * 257 * 4);
    float*     skipA = (float*)carve((size_t)4 * 3 * 128 * 128 * 4);
    float*     skipB = (float*)carve((size_t)4 * 3 * 128 * 128 * 4);

    // ---- mapping network ----
    pixelnorm_kernel<<<4, 256, 0, stream>>>(z, wlat);
    const float mscale = 0.01f / sqrtf(512.f);
    float* mcur = wlat;
    float* mnxt = wtmp;
    for (int i = 0; i < 8; ++i) {
        eqlin_lrelu_kernel<<<8, 256, 0, stream>>>(
            mcur, (const float*)d_in[2 + 2 * i], (const float*)d_in[3 + 2 * i],
            mnxt, mscale, 0.01f);
        float* t = mcur; mcur = mnxt; mnxt = t;
    }
    const float* wsty = mcur;  // [4,512]

    // ---- styled conv launcher ----
    auto run_conv = [&](int L, const float* act_in, int bstride, float* act_out) {
        const CCfg c = g_conv[L];
        const float* W    = (const float*)d_in[18 + 5 * L + 0];
        const float* modw = (const float*)d_in[18 + 5 * L + 1];
        const float* modb = (const float*)d_in[18 + 5 * L + 2];
        const float* nw   = (const float*)d_in[18 + 5 * L + 3];
        const float* actb = (const float*)d_in[18 + 5 * L + 4];
        const float* noise = (const float*)d_in[97 + L];
        const float scale = 1.f / sqrtf((float)(c.cin * 9));

        style_kernel<<<cdiv(4 * c.cin, 256), 256, 0, stream>>>(wsty, modw, modb, sbuf, c.cin);
        wsq_kernel<<<cdiv(c.cout * c.cin, 256), 256, 0, stream>>>(W, wsqb, c.cout * c.cin);
        demod_kernel<<<cdiv(4 * c.cout, 256), 256, 0, stream>>>(sbuf, wsqb, dbuf,
                                                                c.cin, c.cout, scale * scale);
        int HWi = c.Hin * c.Hin;
        modulate_kernel<<<cdiv(4 * c.cin * HWi, 256), 256, 0, stream>>>(
            act_in, sbuf, xm16, c.cin, HWi, bstride, 4 * c.cin * HWi);

        int Hout;
        if (!c.up) {
            wcvt_kernel<<<cdiv(c.cout * c.cin * 9, 256), 256, 0, stream>>>(
                W, a16, c.cout * c.cin * 9, scale);
            Hout = c.Hin;
            int Np = Hout * Hout;
            dim3 g(cdiv(Np, 64), c.cout / 128, 4);
            conv_wmma_kernel<<<g, 256, 0, stream>>>(a16, xm16, dbuf, actb, act_out,
                                                    c.cin, c.cout, c.Hin, 0, 0, 0);
        } else {
            Hout = 2 * c.Hin;
            const size_t CC = (size_t)c.cout * c.cin;
            const size_t phOff[4] = {0, 4 * CC, 6 * CC, 8 * CC};  // ntap 4,2,2,1
            for (int ph = 0; ph < 4; ++ph) {
                int py = ph >> 1, px = ph & 1;
                int ntap = (py ? 1 : 2) * (px ? 1 : 2);
                int tot = (int)(CC * ntap);
                wcvt_phase_kernel<<<cdiv(tot, 256), 256, 0, stream>>>(
                    W, a16 + phOff[ph], c.cin, tot, py, px, scale);
            }
            for (int ph = 0; ph < 4; ++ph) {
                int py = ph >> 1, px = ph & 1;
                int Hp = py ? c.Hin : c.Hin + 1;
                int Wp = px ? c.Hin : c.Hin + 1;
                int Np = Hp * Wp;
                dim3 g(cdiv(Np, 64), c.cout / 128, 4);
                conv_wmma_kernel<<<g, 256, 0, stream>>>(a16 + phOff[ph], xm16, dbuf, actb,
                                                        convT, c.cin, c.cout, c.Hin, 1,
                                                        py, px);
            }
            int tot = 4 * c.cout * Hout * Hout;
            blur_act_kernel<<<cdiv(tot, 256), 256, 0, stream>>>(convT, dbuf, actb, act_out,
                                                                c.cout, Hout, tot);
        }
        int HWo = Hout * Hout;
        int tot = 4 * c.cout * HWo;
        noise_fill_kernel<<<cdiv(tot, 256), 256, 0, stream>>>(noise, nw, actb, act_out,
                                                              c.cout, HWo, tot);
    };

    auto run_rgb = [&](int T, const float* act_in, int H, float* out_rgb) {
        int cin = g_rgb_cin[T];
        const float* W    = (const float*)d_in[73 + 4 * T + 0];
        const float* modw = (const float*)d_in[73 + 4 * T + 1];
        const float* modb = (const float*)d_in[73 + 4 * T + 2];
        const float* bias = (const float*)d_in[73 + 4 * T + 3];
        style_kernel<<<cdiv(4 * cin, 256), 256, 0, stream>>>(wsty, modw, modb, sbuf, cin);
        wrgb_kernel<<<cdiv(4 * 3 * cin, 256), 256, 0, stream>>>(
            W, sbuf, wrgbb, cin, 1.f / sqrtf((float)cin), 4 * 3 * cin);
        int HW = H * H;
        rgb_kernel<<<cdiv(4 * HW, 256), 256, 0, stream>>>(act_in, wrgbb, bias, out_rgb,
                                                          cin, HW, 4 * HW);
    };

    // ---- generator pipeline ----
    run_conv(0, cst, 0 /*broadcast const over batch*/, actA);   // [4,1024,4,4]
    run_rgb(0, actA, 4, skipA);

    float* act = actA;
    float* actO = actB;
    float* skip = skipA;
    int Hc = 4;
    for (int blk = 0; blk < 5; ++blk) {
        int L1 = 1 + 2 * blk, L2 = 2 + 2 * blk;
        const CCfg c1 = g_conv[L1];
        run_conv(L1, act, c1.cin * Hc * Hc, actO);              // upsample conv
        { float* t = act; act = actO; actO = t; }
        Hc *= 2;
        const CCfg c2 = g_conv[L2];
        run_conv(L2, act, c2.cin * Hc * Hc, actO);
        { float* t = act; act = actO; actO = t; }

        float* rgb_dst = (blk == 4) ? (float*)d_out
                                    : (skip == skipA ? skipB : skipA);
        run_rgb(1 + blk, act, Hc, rgb_dst);
        int tot = 4 * 3 * Hc * Hc;
        skip_up_add_kernel<<<cdiv(tot, 256), 256, 0, stream>>>(skip, rgb_dst, Hc / 2, tot);
        skip = rgb_dst;
    }
}
